// FOFE_GRU_89893665505740
// MI455X (gfx1250) — matrix operand order repaired and checked by reading, not compiled, not run on hardware
//
#include <hip/hip_runtime.h>
#include <stdint.h>

#define B_  128
#define S_  128
#define Wn_ 16
#define V_  128
#define H_  512
#define L_  64
#define C2H (2 * H_)

typedef __attribute__((ext_vector_type(16))) __bf16 v16bf;
typedef __attribute__((ext_vector_type(8)))  __bf16 bf16x8;
typedef __attribute__((ext_vector_type(8)))  float  v8f;

// ---------------- WMMA fragment loaders (CDNA5 wave32 layouts) ----------------
// A (16x32 bf16, MxK): lane L -> M = L&15, K = (L>>4)*16 .. +15 contiguous
static __device__ __forceinline__ v16bf load_a_frag(const __bf16* p) {
  return *(const v16bf*)(p);
}
// B (32x16 bf16, KxN): lane L -> N = L&15, K = (L>>4)*16 .. +15 contiguous
static __device__ __forceinline__ v16bf load_b_frag(const __bf16* p) {
  return *(const v16bf*)(p);
}

static __device__ __forceinline__ float sigmoidf_(float x) {
  return 1.0f / (1.0f + __expf(-x));
}

// ---------------- prep kernels ----------------
__global__ void cvt_f32_bf16_kernel(const float* __restrict__ src,
                                    __bf16* __restrict__ dst, int n) {
  int i = blockIdx.x * blockDim.x + threadIdx.x;
  if (i < n) dst[i] = (__bf16)src[i];
}

__global__ void zero_u32_kernel(uint32_t* __restrict__ p, int n) {
  int i = blockIdx.x * blockDim.x + threadIdx.x;
  if (i < n) p[i] = 0u;
}

// ---------------- FOFE: chars (B,S,16) -> x bf16 (B*S, 128) ----------------
// one wave per (b,s) row; lane owns vocab slots {lane, lane+32, lane+64, lane+96}
__global__ void fofe_kernel(const int* __restrict__ chars,
                            const float* __restrict__ forget_p,
                            __bf16* __restrict__ x_bf) {
  int row  = blockIdx.x * 8 + (threadIdx.x >> 5);   // 0 .. B*S-1
  int lane = threadIdx.x & 31;
  const float f = *forget_p;

  int c[Wn_];
#pragma unroll
  for (int w = 0; w < Wn_; ++w) c[w] = chars[(size_t)row * Wn_ + w];

  // cnt[w] = # nonzero chars strictly after w; wt = mask * f^cnt
  float wt[Wn_];
  int cnt = 0;
#pragma unroll
  for (int w = Wn_ - 1; w >= 0; --w) {
    wt[w] = (c[w] != 0) ? __powf(f, (float)cnt) : 0.0f;
    cnt += (c[w] != 0) ? 1 : 0;
  }

  float acc[4] = {0.f, 0.f, 0.f, 0.f};
#pragma unroll
  for (int w = 0; w < Wn_; ++w) {
    int cc = c[w];
    if ((cc & 31) == lane) acc[cc >> 5] += wt[w];
  }
#pragma unroll
  for (int q = 0; q < 4; ++q)
    x_bf[(size_t)row * V_ + q * 32 + lane] = (__bf16)acc[q];
}

// ---------------- fused GRU step (input proj + recurrence + gates) ----------------
// grid = (B/16, H/128, 2 dirs), block = 256 (8 waves).
// wave w owns output columns j0 = by*128 + w*16 .. +15 for its 16 batch rows.
// K loop runs over concatenated [x_t (128) ; h_prev (512)] = 640 with weights
// [W_ih ; W_hh]; n-gate keeps xn / hn accumulators separate.
__global__ void __launch_bounds__(256)
gru_step_kernel(const __bf16* __restrict__ x_bf,
                const int*    __restrict__ lengths,
                const __bf16* __restrict__ Wih_f, const __bf16* __restrict__ Whh_f,
                const float*  __restrict__ bih_f, const float*  __restrict__ bhh_f,
                const __bf16* __restrict__ Wih_b, const __bf16* __restrict__ Whh_b,
                const float*  __restrict__ bih_b, const float*  __restrict__ bhh_b,
                const float*  __restrict__ h32_in_f,  float*  __restrict__ h32_out_f,
                const __bf16* __restrict__ hbf_in_f,  __bf16* __restrict__ hbf_out_f,
                const float*  __restrict__ h32_in_b,  float*  __restrict__ h32_out_b,
                const __bf16* __restrict__ hbf_in_b,  __bf16* __restrict__ hbf_out_b,
                __bf16* __restrict__ concat, int t) {
  const int wave = threadIdx.x >> 5;
  const int lane = threadIdx.x & 31;
  const int dir  = blockIdx.z;
  const int b0   = blockIdx.x * 16;
  const int j0   = blockIdx.y * 128 + wave * 16;

  const __bf16* Wih    = dir ? Wih_b   : Wih_f;
  const __bf16* Whh    = dir ? Whh_b   : Whh_f;
  const float*  bih    = dir ? bih_b   : bih_f;
  const float*  bhh    = dir ? bhh_b   : bhh_f;
  const float*  h32_in = dir ? h32_in_b : h32_in_f;
  float*  h32_out      = dir ? h32_out_b : h32_out_f;
  const __bf16* hbf_in = dir ? hbf_in_b : hbf_in_f;
  __bf16* hbf_out      = dir ? hbf_out_b : hbf_out_f;

  // A-fragment row for this lane
  const int am    = lane & 15;
  const int kbase = (lane >> 4) << 4;
  const int arow_b = b0 + am;
  const int len_a  = lengths[arow_b];
  int sx = dir ? (len_a - 1 - t) : t;
  if (sx < 0) sx = 0;
  const __bf16* xrow = x_bf   + ((size_t)arow_b * S_ + sx) * V_;
  const __bf16* hrow = hbf_in + (size_t)arow_b * H_;

  // B-fragment rows: weight rows for gates r / z / n at this lane's column
  const int jc    = j0 + (lane & 15);
  const int khalf = (lane >> 4) << 4;
  const __bf16* wih_r = Wih + (size_t)(0 * H_ + jc) * V_;
  const __bf16* wih_z = Wih + (size_t)(1 * H_ + jc) * V_;
  const __bf16* wih_n = Wih + (size_t)(2 * H_ + jc) * V_;
  const __bf16* whh_r = Whh + (size_t)(0 * H_ + jc) * H_;
  const __bf16* whh_z = Whh + (size_t)(1 * H_ + jc) * H_;
  const __bf16* whh_n = Whh + (size_t)(2 * H_ + jc) * H_;

  v8f cr  = {}; v8f cz  = {}; v8f cxn = {}; v8f chn = {};

  // x region: K = 0..127  (contributes to r, z, xn)
#pragma unroll
  for (int kc = 0; kc < V_; kc += 32) {
    v16bf a  = load_a_frag(xrow + kc + kbase);
    v16bf br = load_b_frag(wih_r + kc + khalf);
    cr  = __builtin_amdgcn_wmma_f32_16x16x32_bf16(false, a, false, br, (short)0, cr,  false, false);
    v16bf bz = load_b_frag(wih_z + kc + khalf);
    cz  = __builtin_amdgcn_wmma_f32_16x16x32_bf16(false, a, false, bz, (short)0, cz,  false, false);
    v16bf bn = load_b_frag(wih_n + kc + khalf);
    cxn = __builtin_amdgcn_wmma_f32_16x16x32_bf16(false, a, false, bn, (short)0, cxn, false, false);
  }
  // h region: K = 0..511  (contributes to r, z, hn)
#pragma unroll 4
  for (int kc = 0; kc < H_; kc += 32) {
    v16bf a  = load_a_frag(hrow + kc + kbase);
    v16bf br = load_b_frag(whh_r + kc + khalf);
    cr  = __builtin_amdgcn_wmma_f32_16x16x32_bf16(false, a, false, br, (short)0, cr,  false, false);
    v16bf bz = load_b_frag(whh_z + kc + khalf);
    cz  = __builtin_amdgcn_wmma_f32_16x16x32_bf16(false, a, false, bz, (short)0, cz,  false, false);
    v16bf bn = load_b_frag(whh_n + kc + khalf);
    chn = __builtin_amdgcn_wmma_f32_16x16x32_bf16(false, a, false, bn, (short)0, chn, false, false);
  }

  // gate epilogue: C/D layout: element v -> M = v + 8*(lane>=16), N = lane&15
  const int j = jc;
  const float bir = bih[j],          bhr = bhh[j];
  const float biz = bih[H_ + j],     bhz = bhh[H_ + j];
  const float bin = bih[2 * H_ + j], bhn = bhh[2 * H_ + j];

#pragma unroll
  for (int v = 0; v < 8; ++v) {
    const int b = b0 + v + ((lane >> 4) << 3);
    const float r  = sigmoidf_(cr[v] + bir + bhr);
    const float z  = sigmoidf_(cz[v] + biz + bhz);
    const float nn = tanhf((cxn[v] + bin) + r * (chn[v] + bhn));
    const float hp = h32_in[(size_t)b * H_ + j];
    const float hnew = (1.0f - z) * nn + z * hp;
    const int   len  = lengths[b];
    const bool  msk  = (t < len);
    const float hk   = msk ? hnew : hp;
    h32_out[(size_t)b * H_ + j] = hk;
    hbf_out[(size_t)b * H_ + j] = (__bf16)hk;
    if (dir == 0) {
      concat[((size_t)b * S_ + t) * C2H + j] = (__bf16)(msk ? hnew : 0.0f);
    } else if (msk) {
      const int s = len - 1 - t;   // scatter back to original position
      concat[((size_t)b * S_ + s) * C2H + H_ + j] = (__bf16)hnew;
    }
  }
}

// ---------------- final projection: concat (16384 x 1024) @ W_lin^T -> (16384 x 64) --------
__global__ void __launch_bounds__(256)
out_gemm_kernel(const __bf16* __restrict__ concat,
                const __bf16* __restrict__ Wlin,
                const float*  __restrict__ blin,
                float* __restrict__ out) {
  const int wave = threadIdx.x >> 5;
  const int lane = threadIdx.x & 31;
  const int mt = blockIdx.x * 2 + (wave >> 2);
  const int nt = wave & 3;

  const int kbase = (lane >> 4) << 4;
  const int khalf = (lane >> 4) << 4;
  const __bf16* arow = concat + (size_t)(mt * 16 + (lane & 15)) * C2H;
  const __bf16* brow = Wlin   + (size_t)(nt * 16 + (lane & 15)) * C2H;

  v8f c = {};
#pragma unroll 8
  for (int kc = 0; kc < C2H; kc += 32) {
    v16bf a = load_a_frag(arow + kc + kbase);
    v16bf b = load_b_frag(brow + kc + khalf);
    c = __builtin_amdgcn_wmma_f32_16x16x32_bf16(false, a, false, b, (short)0, c, false, false);
  }
  const int n  = nt * 16 + (lane & 15);
  const float bl = blin[n];
#pragma unroll
  for (int v = 0; v < 8; ++v) {
    const int m = mt * 16 + v + ((lane >> 4) << 3);
    out[(size_t)m * L_ + n] = c[v] + bl;
  }
}

// ---------------- host launcher ----------------
extern "C" void kernel_launch(void* const* d_in, const int* in_sizes, int n_in,
                              void* d_out, int out_size, void* d_ws, size_t ws_size,
                              hipStream_t stream) {
  const int*   chars   = (const int*)  d_in[0];
  const int*   lengths = (const int*)  d_in[1];
  const float* forget  = (const float*)d_in[2];
  const float* W_ih_f  = (const float*)d_in[3];
  const float* W_hh_f  = (const float*)d_in[4];
  const float* b_ih_f  = (const float*)d_in[5];
  const float* b_hh_f  = (const float*)d_in[6];
  const float* W_ih_b  = (const float*)d_in[7];
  const float* W_hh_b  = (const float*)d_in[8];
  const float* b_ih_b  = (const float*)d_in[9];
  const float* b_hh_b  = (const float*)d_in[10];
  const float* W_lin   = (const float*)d_in[11];
  const float* b_lin   = (const float*)d_in[12];
  float* out = (float*)d_out;

  // workspace carve-up (256B aligned)
  char* ws = (char*)d_ws;
  size_t cur = 0;
  auto alloc = [&](size_t bytes) {
    void* p = ws + cur;
    cur += (bytes + 255) & ~(size_t)255;
    return p;
  };
  __bf16* x_bf    = (__bf16*)alloc((size_t)B_ * S_ * V_ * 2);
  __bf16* wih_f   = (__bf16*)alloc((size_t)3 * H_ * V_ * 2);
  __bf16* whh_f   = (__bf16*)alloc((size_t)3 * H_ * H_ * 2);
  __bf16* wih_b   = (__bf16*)alloc((size_t)3 * H_ * V_ * 2);
  __bf16* whh_b   = (__bf16*)alloc((size_t)3 * H_ * H_ * 2);
  __bf16* wlin    = (__bf16*)alloc((size_t)L_ * C2H * 2);
  __bf16* concat  = (__bf16*)alloc((size_t)B_ * S_ * C2H * 2);
  float*  h32f[2] = {(float*)alloc((size_t)B_ * H_ * 4), (float*)alloc((size_t)B_ * H_ * 4)};
  float*  h32b[2] = {(float*)alloc((size_t)B_ * H_ * 4), (float*)alloc((size_t)B_ * H_ * 4)};
  __bf16* hbff[2] = {(__bf16*)alloc((size_t)B_ * H_ * 2), (__bf16*)alloc((size_t)B_ * H_ * 2)};
  __bf16* hbfb[2] = {(__bf16*)alloc((size_t)B_ * H_ * 2), (__bf16*)alloc((size_t)B_ * H_ * 2)};

  // 1) convert weights fp32 -> bf16
  auto cvt = [&](const float* s, __bf16* d, int n) {
    cvt_f32_bf16_kernel<<<(n + 255) / 256, 256, 0, stream>>>(s, d, n);
  };
  cvt(W_ih_f, wih_f, 3 * H_ * V_);
  cvt(W_hh_f, whh_f, 3 * H_ * H_);
  cvt(W_ih_b, wih_b, 3 * H_ * V_);
  cvt(W_hh_b, whh_b, 3 * H_ * H_);
  cvt(W_lin,  wlin,  L_ * C2H);

  // 2) zero concat buffer and initial hidden states (buffer 0)
  auto zero = [&](void* p, size_t bytes) {
    int n = (int)(bytes / 4);
    zero_u32_kernel<<<(n + 255) / 256, 256, 0, stream>>>((uint32_t*)p, n);
  };
  zero(concat,  (size_t)B_ * S_ * C2H * 2);
  zero(h32f[0], (size_t)B_ * H_ * 4);
  zero(h32b[0], (size_t)B_ * H_ * 4);
  zero(hbff[0], (size_t)B_ * H_ * 2);
  zero(hbfb[0], (size_t)B_ * H_ * 2);

  // 3) FOFE encoding
  fofe_kernel<<<(B_ * S_) / 8, 256, 0, stream>>>(chars, forget, x_bf);

  // 4) recurrent sweep: one fused step kernel per timestep, both directions
  for (int t = 0; t < S_; ++t) {
    const int pi = t & 1, po = pi ^ 1;
    gru_step_kernel<<<dim3(B_ / 16, H_ / 128, 2), 256, 0, stream>>>(
        x_bf, lengths,
        wih_f, whh_f, b_ih_f, b_hh_f,
        wih_b, whh_b, b_ih_b, b_hh_b,
        h32f[pi], h32f[po], hbff[pi], hbff[po],
        h32b[pi], h32b[po], hbfb[pi], hbfb[po],
        concat, t);
  }

  // 5) output projection
  out_gemm_kernel<<<(B_ * S_) / 32, 256, 0, stream>>>(concat, wlin, b_lin, out);

  (void)in_sizes; (void)n_in; (void)out_size; (void)ws_size;
}
